// Model_62886911148377
// MI455X (gfx1250) — matrix operand order repaired and checked by reading, not compiled, not run on hardware
//
#include <hip/hip_runtime.h>
#include <hip/hip_bf16.h>

#define DEV __device__ __forceinline__

typedef __attribute__((ext_vector_type(16))) __bf16 bf16x16;
typedef __attribute__((ext_vector_type(8)))  float  f32x8;
typedef unsigned short u16;
typedef unsigned int   u32;
typedef int v4i __attribute__((vector_size(16)));

// ---------------- problem constants ----------------
constexpr int cB = 16, cL = 100, cW = 10;
constexpr int cDC = 400, cDH = 450, cDW = 20;
constexpr int cK = 640, cC = 120;
constexpr int cN = cB * cL * cW;          // 16000 candidate spans
constexpr int DIN  = 1270;                // span emb dim
constexpr int DPAD = 1280;                // padded (multiple of 32)
constexpr int HID  = 150, HPAD = 160;     // ffn hidden (padded to 16)
constexpr int NT = HPAD / 16;             // 10 hidden tiles
constexpr int KT = DPAD / 32;             // 40 k tiles
constexpr int P2 = 16384;                 // bitonic sort size >= cN
constexpr float NEGV = -1e30f;

#if defined(__gfx1250__) && __has_builtin(__builtin_amdgcn_global_load_async_to_lds_b128) && __has_builtin(__builtin_amdgcn_s_wait_asynccnt)
#define HAVE_ASYNC_LDS 1
#else
#define HAVE_ASYNC_LDS 0
#endif

// ---------------- bf16 helpers ----------------
DEV u32 as_u32(float f) { union { float f; u32 u; } c; c.f = f; return c.u; }
DEV float as_f32(u32 u) { union { float f; u32 u; } c; c.u = u; return c.f; }
DEV u16 f2bf(float f) {
  u32 x = as_u32(f);
  return (u16)((x + 0x7FFFu + ((x >> 16) & 1u)) >> 16);
}
DEV float bf2f(u16 h) { return as_f32(((u32)h) << 16); }

// ---------------- WMMA fragments (wave32, ISA layouts) ----------------
// 16-bit A fragment (16M x 32K): lane (m = lane&15, kh = lane>>4) holds
// K = [8kh .. 8kh+7] and [16+8kh .. 23+8kh]  -> two contiguous 16B runs.
DEV bf16x16 frag_from_row(const u16* row, int kbase, int kh) {
  union { bf16x16 v; uint4 q[2]; } f;
  f.q[0] = *(const uint4*)(row + kbase + 8 * kh);
  f.q[1] = *(const uint4*)(row + kbase + 16 + 8 * kh);
  return f.v;
}
DEV bf16x16 load_frag_a(const u16* src, int ld, int rowbase, int kbase, int lane) {
  return frag_from_row(src + (size_t)(rowbase + (lane & 15)) * ld, kbase, (lane >> 4) & 1);
}
// B fragment from pre-swizzled weights: frag = nt*KT + kt, lane-contiguous
DEV bf16x16 load_frag_b_swz(const u16* wswz, int nt, int kt, int lane) {
  const u16* p = wswz + ((size_t)(nt * KT + kt)) * 512 + lane * 16;
  union { bf16x16 v; uint4 q[2]; } f;
  f.q[0] = ((const uint4*)p)[0];
  f.q[1] = ((const uint4*)p)[1];
  return f.v;
}
// elementwise bf16 product of two fragments (in registers)
DEV bf16x16 mul_frag(bf16x16 a, bf16x16 b) {
  union { bf16x16 v; u32 w[8]; } ua, ub, uo;
  ua.v = a; ub.v = b;
#pragma unroll
  for (int i = 0; i < 8; ++i) {
    u32 x = ua.w[i], y = ub.w[i];
    float lo = as_f32(x << 16) * as_f32(y << 16);
    float hi = as_f32(x & 0xFFFF0000u) * as_f32(y & 0xFFFF0000u);
    uo.w[i] = (u32)f2bf(lo) | (((u32)f2bf(hi)) << 16);
  }
  return uo.v;
}
DEV f32x8 wmma_bf16(bf16x16 a, bf16x16 b, f32x8 c) {
  return __builtin_amdgcn_wmma_f32_16x16x32_bf16(false, a, false, b, (short)0, c,
                                                 false, false);
}
// C/D 16x16 f32: element r -> (M = r + 8*(lane>>4), N = lane&15)

// ---------------- kernels ----------------
__global__ void k_zero(float* out) { out[0] = 0.f; }

// convert [1270,150] f32 chunk (rows from rowoff) into fragment-swizzled bf16:
// dst[(nt*KT+kt)*512 + lane*16 + idx] = W[kt*32 + 8*(lane>>4) + (idx&7) + 16*(idx>>3)][nt*16 + (lane&15)]
__global__ void k_convw(const float* __restrict__ src, int rowoff, u16* __restrict__ dst) {
  int t = blockIdx.x * 256 + threadIdx.x;
  if (t >= NT * KT * 512) return;
  int frag = t >> 9, e = t & 511;
  int lane = e >> 4, idx = e & 15;
  int nt = frag / KT, kt = frag % KT;
  int k = kt * 32 + 8 * (lane >> 4) + (idx & 7) + 16 * (idx >> 3);
  int n = nt * 16 + (lane & 15);
  float v = (k < DIN && n < HID) ? src[(size_t)(rowoff + k) * HID + n] : 0.f;
  dst[t] = f2bf(v);
}

// span embeddings: one block per span; out bf16 [cN][DPAD]
__global__ void k_span(const float* __restrict__ ctx, const float* __restrict__ head,
                       const float* __restrict__ wemb, const float* __restrict__ wattn,
                       u16* __restrict__ span) {
  const int n = blockIdx.x, tid = threadIdx.x;
  const int LW = cL * cW;
  const int b = n / LW, rr = n % LW, start = rr / cW, wd = rr % cW;
  const int fs = b * cL + start;
  const int feu = fs + wd;
  const int fe = feu < (cB * cL - 1) ? feu : (cB * cL - 1);
  __shared__ float red[256];
  __shared__ float logits[cW];
  for (int tok = 0; tok < cW; ++tok) {
    int ti = fs + tok; if (ti > cB * cL - 1) ti = cB * cL - 1;
    float p = 0.f;
    for (int d = tid; d < cDH; d += 256) p += head[(size_t)ti * cDH + d] * wattn[d];
    red[tid] = p; __syncthreads();
    for (int s = 128; s > 0; s >>= 1) { if (tid < s) red[tid] += red[tid + s]; __syncthreads(); }
    if (tid == 0) logits[tok] = (tok <= wd) ? red[0] : NEGV;
    __syncthreads();
  }
  float mx = NEGV;
#pragma unroll
  for (int tok = 0; tok < cW; ++tok) mx = fmaxf(mx, logits[tok]);
  float aw[cW]; float den = 0.f;
#pragma unroll
  for (int tok = 0; tok < cW; ++tok) { aw[tok] = __expf(logits[tok] - mx); den += aw[tok]; }
  const float inv = 1.f / den;

  u16* out = span + (size_t)n * DPAD;
  for (int d = tid; d < cDC; d += 256) {
    out[d]       = f2bf(ctx[(size_t)fs * cDC + d]);
    out[cDC + d] = f2bf(ctx[(size_t)fe * cDC + d]);
  }
  for (int d = tid; d < cDW; d += 256) out[2 * cDC + d] = f2bf(wemb[(size_t)wd * cDW + d]);
  for (int d = tid; d < cDH; d += 256) {
    float s = 0.f;
#pragma unroll
    for (int tok = 0; tok < cW; ++tok) {
      int ti = fs + tok; if (ti > cB * cL - 1) ti = cB * cL - 1;
      s += aw[tok] * head[(size_t)ti * cDH + d];
    }
    out[2 * cDC + cDW + d] = f2bf(s * inv);
  }
  for (int d = tid; d < DPAD - DIN; d += 256) out[DIN + d] = 0;
}

// unary mention scores via WMMA; writes masked scores + sort keys
__global__ void k_unary(const u16* __restrict__ span, const u16* __restrict__ WU,
                        const float* __restrict__ bu1, const float* __restrict__ wu2,
                        const float* __restrict__ bu2, const int* __restrict__ text_len,
                        float* __restrict__ scores, float* __restrict__ skey,
                        int* __restrict__ sval) {
  const int wave = blockIdx.x * (blockDim.x >> 5) + (threadIdx.x >> 5); // 0..999
  const int lane = threadIdx.x & 31;
  const int rowbase = wave * 16;
  float partial[8];
#pragma unroll
  for (int r = 0; r < 8; ++r) partial[r] = 0.f;
  for (int nt = 0; nt < NT; ++nt) {
    f32x8 acc = {0.f, 0.f, 0.f, 0.f, 0.f, 0.f, 0.f, 0.f};
    for (int kt = 0; kt < KT; ++kt) {
      bf16x16 af = load_frag_a(span, DPAD, rowbase, kt * 32, lane);
      bf16x16 bf = load_frag_b_swz(WU, nt, kt, lane);
      acc = wmma_bf16(af, bf, acc);
    }
    int ng = nt * 16 + (lane & 15);
    if (ng < HID) {
      float bias = bu1[ng], wv = wu2[ng];
#pragma unroll
      for (int r = 0; r < 8; ++r) partial[r] += fmaxf(acc[r] + bias, 0.f) * wv;
    }
  }
#pragma unroll
  for (int r = 0; r < 8; ++r) {
    float p = partial[r];
    p += __shfl_xor(p, 1, 32); p += __shfl_xor(p, 2, 32);
    p += __shfl_xor(p, 4, 32); p += __shfl_xor(p, 8, 32);
    partial[r] = p;
  }
  if ((lane & 15) == 0) {
    const int mbase = rowbase + ((lane >> 4) ? 8 : 0);
#pragma unroll
    for (int r = 0; r < 8; ++r) {
      int row = mbase + r;
      float sc = partial[r] + bu2[0];
      int b = row / (cL * cW), rr = row % (cL * cW), st = rr / cW, wd = rr % cW;
      bool ok = (st + wd) < text_len[b];
      float m = ok ? sc : NEGV;
      scores[row] = m;
      skey[row] = -m;
      sval[row] = row;
    }
  }
}

__global__ void k_sortpad(float* skey, int* sval) {
  int t = cN + blockIdx.x * 256 + threadIdx.x;
  if (t < P2) { skey[t] = 3.0e38f; sval[t] = -1; }
}

__global__ void k_bitonic(float* __restrict__ key, int* __restrict__ val, int j, int k) {
  int i = blockIdx.x * blockDim.x + threadIdx.x;
  int ixj = i ^ j;
  if (ixj > i && ixj < P2) {
    bool up = ((i & k) == 0);
    float ki = key[i], kj = key[ixj];
    if ((ki > kj) == up) {
      key[i] = kj; key[ixj] = ki;
      int t = val[i]; val[i] = val[ixj]; val[ixj] = t;
    }
  }
}

// resort the top-640 by (start,end); single block, LDS bitonic on 1024
__global__ void k_sortspans(const int* __restrict__ sval, int* __restrict__ idxout) {
  __shared__ int k2[1024];
  __shared__ int v2[1024];
  const int t = threadIdx.x;
  int key = 0x7FFFFFFF, val = -1;
  if (t < cK) {
    int id = sval[t];
    int b = id / (cL * cW), rr = id % (cL * cW), st = rr / cW, wd = rr % cW;
    int fs = b * cL + st, fe = fs + wd;
    key = fs * (2 * cB * cL) + fe;
    val = id;
  }
  k2[t] = key; v2[t] = val; __syncthreads();
  for (int kk = 2; kk <= 1024; kk <<= 1)
    for (int j = kk >> 1; j > 0; j >>= 1) {
      int ixj = t ^ j;
      if (ixj > t) {
        bool up = ((t & kk) == 0);
        if ((k2[t] > k2[ixj]) == up) {
          int a = k2[t]; k2[t] = k2[ixj]; k2[ixj] = a;
          int b2 = v2[t]; v2[t] = v2[ixj]; v2[ixj] = b2;
        }
      }
      __syncthreads();
    }
  if (t < cK) idxout[t] = v2[t];
}

__global__ void k_gather(const u16* __restrict__ span, const float* __restrict__ scores,
                         const int* __restrict__ idx, const int* __restrict__ doc_id,
                         const int* __restrict__ cluster, u16* __restrict__ memb,
                         float* __restrict__ msc, int* __restrict__ mdoc,
                         int* __restrict__ mc) {
  const int i = blockIdx.x;
  const int id = idx[i];
  const uint4* src = (const uint4*)(span + (size_t)id * DPAD);
  uint4* dst = (uint4*)(memb + (size_t)i * DPAD);
  for (int q = threadIdx.x; q < DPAD / 8; q += blockDim.x) dst[q] = src[q];
  if (threadIdx.x == 0) {
    msc[i] = scores[id];
    mdoc[i] = doc_id[id / (cL * cW)];
    mc[i] = cluster[id];
  }
}

// U = memb @ Wt, V = memb @ Wa  (f32 out [cK][HPAD])
__global__ void k_uv(const u16* __restrict__ memb, const u16* __restrict__ WT,
                     const u16* __restrict__ WA, float* __restrict__ Uo,
                     float* __restrict__ Vo) {
  const int wave = blockIdx.x * (blockDim.x >> 5) + (threadIdx.x >> 5); // 0..79
  const int lane = threadIdx.x & 31;
  const u16* Wm = (wave < 40) ? WT : WA;
  float* out = (wave < 40) ? Uo : Vo;
  const int rowbase = (wave % 40) * 16;
  for (int nt = 0; nt < NT; ++nt) {
    f32x8 acc = {0.f, 0.f, 0.f, 0.f, 0.f, 0.f, 0.f, 0.f};
    for (int kt = 0; kt < KT; ++kt) {
      bf16x16 af = load_frag_a(memb, DPAD, rowbase, kt * 32, lane);
      bf16x16 bf = load_frag_b_swz(Wm, nt, kt, lane);
      acc = wmma_bf16(af, bf, acc);
    }
    int n = nt * 16 + (lane & 15);
    int mof = (lane >> 4) * 8;
#pragma unroll
    for (int r = 0; r < 8; ++r) out[(size_t)(rowbase + mof + r) * HPAD + n] = acc[r];
  }
}

// antecedent pair scores: block = mention i, wave = 16-antecedent tile
__global__ void k_pair(const u16* __restrict__ memb, const u16* __restrict__ WP,
                       const float* __restrict__ U, const float* __restrict__ V,
                       const float* __restrict__ ba1, const float* __restrict__ wa2,
                       const float* __restrict__ ba2, const float* __restrict__ msc,
                       const int* __restrict__ mdoc, float* __restrict__ ant) {
  __shared__ __align__(16) u16 ldsT[DPAD];       // mention-i row (2.5 KB)
  __shared__ __align__(16) u16 ldsB[KT * 512];   // current nt's 40 B fragments (40 KB)
  const int i = blockIdx.x;
  const int tid = threadIdx.x;
  const int lane = tid & 31;
  const int mtile = tid >> 5;  // 0..7, covers j = mtile*16 .. +15

  // stage mention-i row into LDS
  {
    const uint4* src = (const uint4*)(memb + (size_t)i * DPAD);
    uint4* dst = (uint4*)ldsT;
    for (int q = tid; q < DPAD / 8; q += 256) dst[q] = src[q];
  }

  const int jA = mtile * 16 + (lane & 15);  // pair row this lane feeds into A
  const int rawA = i - 1 - jA;
  const int antA = rawA > 0 ? rawA : 0;
  const u16* arow = memb + (size_t)antA * DPAD;
  const int khA = (lane >> 4) & 1;

  float partial[8];
#pragma unroll
  for (int r = 0; r < 8; ++r) partial[r] = 0.f;

  for (int nt = 0; nt < NT; ++nt) {
    __syncthreads();  // previous tile fully consumed (also orders ldsT staging)
    {
      const uint4* src = (const uint4*)(WP + (size_t)nt * KT * 512);
      uint4* dst = (uint4*)ldsB;
      for (int q = tid; q < KT * 512 / 8; q += 256) {
#if HAVE_ASYNC_LDS
        __builtin_amdgcn_global_load_async_to_lds_b128((v4i*)(src + q),
                                                       (v4i*)(dst + q), 0, 0);
#else
        dst[q] = src[q];
#endif
      }
#if HAVE_ASYNC_LDS
      __builtin_amdgcn_s_wait_asynccnt(0);
#endif
    }
    __syncthreads();

    f32x8 acc = {0.f, 0.f, 0.f, 0.f, 0.f, 0.f, 0.f, 0.f};
    for (int kt = 0; kt < KT; ++kt) {
      bf16x16 tv = frag_from_row(ldsT, kt * 32, khA);
      bf16x16 av = frag_from_row(arow, kt * 32, khA);
      bf16x16 bb;
      {
        const u16* p = ldsB + (size_t)kt * 512 + lane * 16;
        union { bf16x16 v; uint4 q[2]; } f;
        f.q[0] = ((const uint4*)p)[0];
        f.q[1] = ((const uint4*)p)[1];
        bb = f.v;
      }
      acc = wmma_bf16(mul_frag(tv, av), bb, acc);
    }
    int ng = nt * 16 + (lane & 15);
    if (ng < HID) {
      float bias = ba1[ng], wv = wa2[ng], ui = U[(size_t)i * HPAD + ng];
#pragma unroll
      for (int r = 0; r < 8; ++r) {
        int m2 = r + 8 * (lane >> 4);
        int j = mtile * 16 + m2;
        int raw = i - 1 - j;
        int antj = raw > 0 ? raw : 0;
        float hv = fmaxf(acc[r] + ui + V[(size_t)antj * HPAD + ng] + bias, 0.f);
        partial[r] += hv * wv;
      }
    }
  }
#pragma unroll
  for (int r = 0; r < 8; ++r) {
    float p = partial[r];
    p += __shfl_xor(p, 1, 32); p += __shfl_xor(p, 2, 32);
    p += __shfl_xor(p, 4, 32); p += __shfl_xor(p, 8, 32);
    partial[r] = p;
  }
  if ((lane & 15) == 0) {
#pragma unroll
    for (int r = 0; r < 8; ++r) {
      int m2 = r + 8 * (lane >> 4);
      int j = mtile * 16 + m2;
      if (j < cC) {
        int raw = i - 1 - j;
        int antj = raw > 0 ? raw : 0;
        bool am = (raw >= 0) && (mdoc[i] == mdoc[antj]);
        float sc = am ? (partial[r] + ba2[0] + msc[i] + msc[antj]) : NEGV;
        ant[(size_t)i * (1 + cC) + 1 + j] = sc;
      }
    }
  }
  if (tid == 0) ant[(size_t)i * (1 + cC)] = 0.f;
}

// per-row dual logsumexp
__global__ void k_loss(const float* __restrict__ ant, const int* __restrict__ mc,
                       const int* __restrict__ mdoc, float* __restrict__ lossrow) {
  const int i = blockIdx.x, t = threadIdx.x;  // 128 threads
  __shared__ float s[128];
  __shared__ int si[128];
  float score = NEGV;
  int pw = 0;
  if (t < 1 + cC) {
    score = ant[(size_t)i * (1 + cC) + t];
    if (t >= 1) {
      int j = t - 1, raw = i - 1 - j;
      int antj = raw > 0 ? raw : 0;
      int am = (raw >= 0) && (mdoc[i] == mdoc[antj]);
      pw = am && (mc[antj] == mc[i]) && (mc[i] > 0);
    }
  }
  si[t] = pw; __syncthreads();
  for (int st2 = 64; st2 > 0; st2 >>= 1) { if (t < st2) si[t] |= si[t + st2]; __syncthreads(); }
  const int anyp = si[0]; __syncthreads();
  float gold = NEGV;
  if (t == 0) gold = anyp ? NEGV : score;          // dummy label, score[0] == 0
  else if (t < 1 + cC && pw) gold = score;

  s[t] = score; __syncthreads();
  for (int st2 = 64; st2 > 0; st2 >>= 1) { if (t < st2) s[t] = fmaxf(s[t], s[t + st2]); __syncthreads(); }
  const float mx = s[0]; __syncthreads();
  s[t] = __expf(score - mx); __syncthreads();
  for (int st2 = 64; st2 > 0; st2 >>= 1) { if (t < st2) s[t] += s[t + st2]; __syncthreads(); }
  const float lseS = mx + __logf(s[0]); __syncthreads();

  s[t] = gold; __syncthreads();
  for (int st2 = 64; st2 > 0; st2 >>= 1) { if (t < st2) s[t] = fmaxf(s[t], s[t + st2]); __syncthreads(); }
  const float mg = s[0]; __syncthreads();
  s[t] = __expf(gold - mg); __syncthreads();
  for (int st2 = 64; st2 > 0; st2 >>= 1) { if (t < st2) s[t] += s[t + st2]; __syncthreads(); }
  const float lseG = mg + __logf(s[0]);
  if (t == 0) lossrow[i] = lseS - lseG;
}

__global__ void k_reduce(const float* __restrict__ lr, float* __restrict__ out) {
  __shared__ float s[256];
  float a = 0.f;
  for (int i = threadIdx.x; i < cK; i += 256) a += lr[i];
  s[threadIdx.x] = a; __syncthreads();
  for (int st = 128; st > 0; st >>= 1) { if (threadIdx.x < st) s[threadIdx.x] += s[threadIdx.x + st]; __syncthreads(); }
  if (threadIdx.x == 0) out[0] = s[0];
}

// ---------------- host launcher ----------------
extern "C" void kernel_launch(void* const* d_in, const int* in_sizes, int n_in,
                              void* d_out, int out_size, void* d_ws, size_t ws_size,
                              hipStream_t stream) {
  (void)in_sizes; (void)n_in; (void)out_size; (void)ws_size;
  const float* ctx      = (const float*)d_in[0];
  const float* head     = (const float*)d_in[1];
  const float* wemb     = (const float*)d_in[2];
  const float* wattn    = (const float*)d_in[3];
  const float* Wu1      = (const float*)d_in[4];
  const float* bu1      = (const float*)d_in[5];
  const float* wu2      = (const float*)d_in[6];
  const float* bu2      = (const float*)d_in[7];
  const float* Wa1      = (const float*)d_in[8];
  const float* ba1      = (const float*)d_in[9];
  const float* wa2      = (const float*)d_in[10];
  const float* ba2      = (const float*)d_in[11];
  const int* text_len   = (const int*)d_in[12];
  const int* doc_id     = (const int*)d_in[13];
  const int* cluster    = (const int*)d_in[14];
  float* out = (float*)d_out;

  char* w = (char*)d_ws;
  size_t off = 0;
  auto take = [&](size_t bytes) -> char* {
    char* p = w + off;
    off += (bytes + 255) & ~(size_t)255;
    return p;
  };
  u16*   SPAN   = (u16*)  take((size_t)cN * DPAD * sizeof(u16));
  u16*   WU     = (u16*)  take((size_t)NT * KT * 512 * sizeof(u16));
  u16*   WT     = (u16*)  take((size_t)NT * KT * 512 * sizeof(u16));
  u16*   WAm    = (u16*)  take((size_t)NT * KT * 512 * sizeof(u16));
  u16*   WP     = (u16*)  take((size_t)NT * KT * 512 * sizeof(u16));
  float* SCORES = (float*)take((size_t)cN * sizeof(float));
  float* SKEY   = (float*)take((size_t)P2 * sizeof(float));
  int*   SVAL   = (int*)  take((size_t)P2 * sizeof(int));
  int*   IDX    = (int*)  take((size_t)cK * sizeof(int));
  u16*   MEMB   = (u16*)  take((size_t)cK * DPAD * sizeof(u16));
  float* MSC    = (float*)take((size_t)cK * sizeof(float));
  int*   MDOC   = (int*)  take((size_t)cK * sizeof(int));
  int*   MC     = (int*)  take((size_t)cK * sizeof(int));
  float* Ubuf   = (float*)take((size_t)cK * HPAD * sizeof(float));
  float* Vbuf   = (float*)take((size_t)cK * HPAD * sizeof(float));
  float* ANT    = (float*)take((size_t)cK * (1 + cC) * sizeof(float));
  float* LROW   = (float*)take((size_t)cK * sizeof(float));

  k_zero<<<1, 1, 0, stream>>>(out);

  // fragment-swizzled bf16 weights
  const int convBlocks = (NT * KT * 512 + 255) / 256;
  k_convw<<<convBlocks, 256, 0, stream>>>(Wu1, 0,       WU);
  k_convw<<<convBlocks, 256, 0, stream>>>(Wa1, 0,       WT);
  k_convw<<<convBlocks, 256, 0, stream>>>(Wa1, DIN,     WAm);
  k_convw<<<convBlocks, 256, 0, stream>>>(Wa1, 2 * DIN, WP);

  // span embeddings
  k_span<<<cN, 256, 0, stream>>>(ctx, head, wemb, wattn, SPAN);

  // unary mention scores (1000 tiles of 16 rows, 8 waves/block)
  k_unary<<<cN / 16 / 8, 256, 0, stream>>>(SPAN, WU, bu1, wu2, bu2, text_len,
                                           SCORES, SKEY, SVAL);

  // top-K via global bitonic sort (ascending on -score)
  k_sortpad<<<2, 256, 0, stream>>>(SKEY, SVAL);
  for (int k = 2; k <= P2; k <<= 1)
    for (int j = k >> 1; j > 0; j >>= 1)
      k_bitonic<<<P2 / 256, 256, 0, stream>>>(SKEY, SVAL, j, k);
  k_sortspans<<<1, 1024, 0, stream>>>(SVAL, IDX);

  // gather selected mentions
  k_gather<<<cK, 256, 0, stream>>>(SPAN, SCORES, IDX, doc_id, cluster,
                                   MEMB, MSC, MDOC, MC);

  // U/V factored projections (80 waves)
  k_uv<<<10, 256, 0, stream>>>(MEMB, WT, WAm, Ubuf, Vbuf);

  // antecedent pair scores
  k_pair<<<cK, 256, 0, stream>>>(MEMB, WP, Ubuf, Vbuf, ba1, wa2, ba2,
                                 MSC, MDOC, ANT);

  // marginal log-likelihood loss
  k_loss<<<cK, 128, 0, stream>>>(ANT, MC, MDOC, LROW);
  k_reduce<<<1, 256, 0, stream>>>(LROW, out);
}
